// MSAColumnAttention_33681133535498
// MI455X (gfx1250) — matrix operand order repaired and checked by reading, not compiled, not run on hardware
//
#include <hip/hip_runtime.h>
#include <hip/hip_bf16.h>
#include <math.h>
#include <stdint.h>

typedef __attribute__((ext_vector_type(16))) _Float16 v16h;
typedef __attribute__((ext_vector_type(8)))  float    v8f;
typedef __attribute__((ext_vector_type(2)))  float    v2f;

#define S_LEN 256
#define I_LEN 384
#define C_DIM 32
#define NHEAD 8
#define CHD   4
#define NPOS  (S_LEN * I_LEN)   /* 98304 flattened (s,i) positions */
#define TILES_PER_SROW (I_LEN / 16)   /* 24: a 16-pos tile never crosses an s row */

static __device__ __forceinline__ v8f wmma_f16(v16h a, v16h b, v8f c) {
  return __builtin_amdgcn_wmma_f32_16x16x32_f16(false, a, false, b, (short)0, c, false, false);
}

// ---------------------------------------------------------------------------
// Kernel 1: LayerNorm + fused Q/K/V/Gate projections.
// One wave per 16-position tile; lane L and L+16 redundantly LayerNorm row L
// so the f16 A-fragment (contiguous k = 16*hx + s order) packs with no cross-lane
// traffic. tile is forced scalar so s/i decomposition is SALU and fragment
// stores are one base + immediate offsets.
// ---------------------------------------------------------------------------
__global__ void __launch_bounds__(256)
k_ln_qkvg(const float* __restrict__ m, const float* __restrict__ lnw, const float* __restrict__ lnb,
          const float* __restrict__ Wq, const float* __restrict__ Wk,
          const float* __restrict__ Wv, const float* __restrict__ Wg, const float* __restrict__ bg,
          float* __restrict__ qo, float* __restrict__ ko, float* __restrict__ vo,
          float* __restrict__ gateo)
{
  const int lane = threadIdx.x & 31;
  const int tile = __builtin_amdgcn_readfirstlane((blockIdx.x * blockDim.x + threadIdx.x) >> 5);
  const int lm = lane & 15, hx = lane >> 4;
  const int row = tile * 16 + lm;

  const int s_row  = tile / TILES_PER_SROW;                 // SALU div
  const int i_base = (tile - s_row * TILES_PER_SROW) * 16;  // i never wraps inside a tile

  __builtin_prefetch(m + (size_t)(row + 16) * C_DIM, 0, 0); // next tile (speculative)

  float x[C_DIM];
  {
    const float4* mr = (const float4*)(m + (size_t)row * C_DIM);
    float mean = 0.f;
#pragma unroll
    for (int j = 0; j < 8; ++j) {
      float4 t = mr[j];
      x[4*j+0]=t.x; x[4*j+1]=t.y; x[4*j+2]=t.z; x[4*j+3]=t.w;
      mean += t.x + t.y + t.z + t.w;
    }
    mean *= (1.f / C_DIM);
    float var = 0.f;
#pragma unroll
    for (int c = 0; c < C_DIM; ++c) { float d = x[c] - mean; var += d * d; }
    var *= (1.f / C_DIM);
    const float rs = rsqrtf(var + 1e-5f);
#pragma unroll
    for (int c = 0; c < C_DIM; ++c) x[c] = (x[c] - mean) * rs * lnw[c] + lnb[c];
  }

  // A fragment: lanes 0-15 hold K=0..15; lanes 16-31 hold K=16..31
  v16h a;
  { const int kb = hx * 16;
#pragma unroll
    for (int j = 0; j < 8; ++j) { a[j] = (_Float16)x[kb + j]; a[8 + j] = (_Float16)x[kb + 8 + j]; } }

  // B fragment (32x16): B[k][n] = W[nbase+n][k]; lane col = lm, K-base = hx*16
  auto loadB = [&](const float* __restrict__ W, int nbase) -> v16h {
    const float4* p = (const float4*)(W + (nbase + lm) * C_DIM + hx * 16);
    v16h b;
#pragma unroll
    for (int j = 0; j < 4; ++j) {
      float4 t = p[j];
      b[4*j+0]=(_Float16)t.x; b[4*j+1]=(_Float16)t.y; b[4*j+2]=(_Float16)t.z; b[4*j+3]=(_Float16)t.w;
    }
    return b;
  };

  v8f z = {};
  v8f q0 = wmma_f16(a, loadB(Wq, 0), z), q1 = wmma_f16(a, loadB(Wq, 16), z);
  v8f k0 = wmma_f16(a, loadB(Wk, 0), z), k1 = wmma_f16(a, loadB(Wk, 16), z);
  v8f v0 = wmma_f16(a, loadB(Wv, 0), z), v1 = wmma_f16(a, loadB(Wv, 16), z);
  v8f g0 = wmma_f16(a, loadB(Wg, 0), z), g1 = wmma_f16(a, loadB(Wg, 16), z);

  // Scatter to [i][h][s][d]; D layout: VGPR r -> row r (lanes 0-15) / 8+r (16-31)
  auto scat = [&](float* __restrict__ dst, const v8f& c, int nbase) {
    const int col = nbase + lm, h = col >> 2, d = col & 3;
    float* base = dst + ((size_t)(i_base + hx * 8) * NHEAD + h) * (S_LEN * CHD)
                      + s_row * CHD + d;
#pragma unroll
    for (int r = 0; r < 8; ++r)
      base[(size_t)r * (NHEAD * S_LEN * CHD)] = c[r];   // +32KB per row: imm offsets
  };
  scat(qo, q0, 0); scat(qo, q1, 16);
  scat(ko, k0, 0); scat(ko, k1, 16);
  scat(vo, v0, 0); scat(vo, v1, 16);

  { const float b0 = bg[lm], b1 = bg[16 + lm];
    float* gbase = gateo + (size_t)(tile * 16 + hx * 8) * C_DIM;
#pragma unroll
    for (int r = 0; r < 8; ++r) {
      gbase[r * C_DIM + lm]      = 1.f / (1.f + __expf(-(g0[r] + b0)));
      gbase[r * C_DIM + 16 + lm] = 1.f / (1.f + __expf(-(g1[r] + b1)));
    }
  }
}

// ---------------------------------------------------------------------------
// Kernel 2: attention over the MSA-sequence axis, one (i,h) head per block.
// q/k/v staged into LDS with GLOBAL_LOAD_ASYNC_TO_LDS_B128 (ASYNCcnt path);
// each wave keeps a 16x256 fp32 score strip in LDS, so scores never touch HBM.
// Scores: V_WMMA_F32_16X16X4_F32 (K=CH=4, exact fp32). p@v: 8 chained f16
// WMMAs (K=256). q/k/v (37 MB total) stay L2-resident across the 3072 blocks.
// ---------------------------------------------------------------------------
__global__ void __launch_bounds__(256)
k_attn(const float* __restrict__ qi, const float* __restrict__ ki, const float* __restrict__ vi,
       const unsigned char* __restrict__ msk, float* __restrict__ oo)
{
  extern __shared__ char smem[];
  float* kbuf = (float*)(smem);          // [256][4] f32 @ 0
  float* qbuf = (float*)(smem + 4096);   // [256][4] f32 (unscaled)
  float* vbuf = (float*)(smem + 8192);   // [256][4] f32
  float* sc   = (float*)(smem + 12288);  // [8 waves][16][256] f32

  const int i = blockIdx.x >> 3, h = blockIdx.x & 7;
  const int headoff = (i * NHEAD + h) * S_LEN * CHD;
  const int tid = threadIdx.x;

  { // async global->LDS staging: one B128 (16B) per thread per tensor
    const unsigned ldsk = (unsigned)(tid * 16);
    const unsigned ldsq = (unsigned)(4096 + tid * 16);
    const unsigned ldsv = (unsigned)(8192 + tid * 16);
    const uint64_t gk = (uint64_t)(uintptr_t)(ki + headoff) + (uint64_t)tid * 16u;
    const uint64_t gq = (uint64_t)(uintptr_t)(qi + headoff) + (uint64_t)tid * 16u;
    const uint64_t gv = (uint64_t)(uintptr_t)(vi + headoff) + (uint64_t)tid * 16u;
    asm volatile("global_load_async_to_lds_b128 %0, %1, off" :: "v"(ldsk), "v"(gk) : "memory");
    asm volatile("global_load_async_to_lds_b128 %0, %1, off" :: "v"(ldsq), "v"(gq) : "memory");
    asm volatile("global_load_async_to_lds_b128 %0, %1, off" :: "v"(ldsv), "v"(gv) : "memory");
    asm volatile("s_wait_asynccnt 0x0" ::: "memory");
  }
  __syncthreads();

  const int wv = tid >> 5, lane = tid & 31;
  const int lm = lane & 15, hx = lane >> 4;
  float* strip = sc + wv * 16 * S_LEN;

  for (int t = wv; t < 16; t += 8) {              // 2 q-tiles per wave, uniform
    v2f a;                                        // f32 A 16x4: VGPR j holds K = j + 2*hx
    a[0] = 0.5f * qbuf[(t*16 + lm)*CHD + 2*hx + 0];   // fold 1/sqrt(CH)
    a[1] = 0.5f * qbuf[(t*16 + lm)*CHD + 2*hx + 1];

#pragma unroll
    for (int kt = 0; kt < 16; ++kt) {
      v2f b;                                      // f32 B 4x16: B[k][n] = K[n_global][k]
      b[0] = kbuf[(kt*16 + lm)*CHD + 2*hx + 0];
      b[1] = kbuf[(kt*16 + lm)*CHD + 2*hx + 1];
      v8f c = {};
      c = __builtin_amdgcn_wmma_f32_16x16x4_f32(false, a, false, b, (short)0, c, false, false);
      const int kcol = kt*16 + lm;
      const float keep = msk[kcol * I_LEN + i] ? 1.f : 0.f;
#pragma unroll
      for (int r = 0; r < 8; ++r)
        strip[(hx*8 + r) * S_LEN + kcol] = keep * c[r] + (1.f - keep) * -1e9f;
    }
    __builtin_amdgcn_wave_barrier();   // same-wave LDS is in-order; fence the scheduler

    { // softmax of row lm; lane pair (lane, lane^16) split the 256 cols
      float4* rp = (float4*)(strip + lm * S_LEN) + hx * 32;
      float mx = -3.0e38f;
#pragma unroll 4
      for (int c4 = 0; c4 < 32; ++c4) {
        float4 tq = rp[c4];
        mx = fmaxf(mx, fmaxf(fmaxf(tq.x, tq.y), fmaxf(tq.z, tq.w)));
      }
      mx = fmaxf(mx, __shfl_xor(mx, 16, 32));
      float sum = 0.f;
#pragma unroll 4
      for (int c4 = 0; c4 < 32; ++c4) {
        float4 tq = rp[c4];
        tq.x = __expf(tq.x - mx); tq.y = __expf(tq.y - mx);
        tq.z = __expf(tq.z - mx); tq.w = __expf(tq.w - mx);
        rp[c4] = tq; sum += tq.x + tq.y + tq.z + tq.w;
      }
      sum += __shfl_xor(sum, 16, 32);
      const float inv = 1.f / sum;
#pragma unroll 4
      for (int c4 = 0; c4 < 32; ++c4) {
        float4 tq = rp[c4];
        tq.x*=inv; tq.y*=inv; tq.z*=inv; tq.w*=inv; rp[c4]=tq;
      }
    }
    __builtin_amdgcn_wave_barrier();

    // O(16x4) = P(16x256) @ V(256x4): 8 chained f16 WMMAs (K=32 each)
    v8f acc = {};
#pragma unroll
    for (int kc = 0; kc < 8; ++kc) {
      v16h pa;
      { const int kb = hx * 16;
        const float* pr = strip + lm * S_LEN + kc * 32;
#pragma unroll
        for (int j = 0; j < 8; ++j) {
          pa[j]   = (_Float16)pr[kb + j];
          pa[8+j] = (_Float16)pr[kb + 8 + j];
        } }
      v16h vb;
      { const int kb2 = hx * 16;
#pragma unroll
        for (int j = 0; j < 16; ++j)
          vb[j] = (lm < CHD) ? (_Float16)vbuf[(kc*32 + kb2 + j)*CHD + lm] : (_Float16)0.f; }
      acc = wmma_f16(pa, vb, acc);
    }
    if (lm < CHD) {
      float* obase = oo + ((size_t)(t*16 + hx*8) * I_LEN + i) * C_DIM + h * CHD + lm;
#pragma unroll
      for (int r = 0; r < 8; ++r)
        obase[(size_t)r * (I_LEN * C_DIM)] = acc[r];   // +48KB per row: imm offsets
    }
  }
}

// ---------------------------------------------------------------------------
// Kernel 3: out = (o * gate) @ Wo^T + bo, same 16-row-tile WMMA scheme.
// ---------------------------------------------------------------------------
__global__ void __launch_bounds__(256)
k_gate_out(const float* __restrict__ oi, const float* __restrict__ gatei,
           const float* __restrict__ Wo, const float* __restrict__ bo, float* __restrict__ out)
{
  const int lane = threadIdx.x & 31;
  const int tile = __builtin_amdgcn_readfirstlane((blockIdx.x * blockDim.x + threadIdx.x) >> 5);
  const int lm = lane & 15, hx = lane >> 4;
  const int row = tile * 16 + lm;

  __builtin_prefetch(oi + (size_t)(row + 16) * C_DIM, 0, 0);

  float x[C_DIM];
  { const float4* orow = (const float4*)(oi + (size_t)row * C_DIM);
    const float4* grow = (const float4*)(gatei + (size_t)row * C_DIM);
#pragma unroll
    for (int j = 0; j < 8; ++j) {
      float4 t = orow[j], g = grow[j];
      x[4*j+0]=t.x*g.x; x[4*j+1]=t.y*g.y; x[4*j+2]=t.z*g.z; x[4*j+3]=t.w*g.w;
    } }

  v16h a;
  { const int kb = hx * 16;
#pragma unroll
    for (int j = 0; j < 8; ++j) { a[j]=(_Float16)x[kb+j]; a[8+j]=(_Float16)x[kb+8+j]; } }

  auto loadB = [&](const float* __restrict__ W, int nbase) -> v16h {
    const float4* p = (const float4*)(W + (nbase + lm) * C_DIM + hx * 16);
    v16h b;
#pragma unroll
    for (int j = 0; j < 4; ++j) {
      float4 t = p[j];
      b[4*j+0]=(_Float16)t.x; b[4*j+1]=(_Float16)t.y; b[4*j+2]=(_Float16)t.z; b[4*j+3]=(_Float16)t.w;
    }
    return b;
  };

  v8f z = {};
  v8f y0 = wmma_f16(a, loadB(Wo, 0), z);
  v8f y1 = wmma_f16(a, loadB(Wo, 16), z);
  const float b0 = bo[lm], b1 = bo[16 + lm];
  float* obase = out + (size_t)(tile * 16 + hx * 8) * C_DIM;
#pragma unroll
  for (int r = 0; r < 8; ++r) {
    obase[r * C_DIM + lm]      = y0[r] + b0;
    obase[r * C_DIM + 16 + lm] = y1[r] + b1;
  }
}

// ---------------------------------------------------------------------------
extern "C" void kernel_launch(void* const* d_in, const int* in_sizes, int n_in,
                              void* d_out, int out_size, void* d_ws, size_t ws_size,
                              hipStream_t stream)
{
  const float*         m    = (const float*)d_in[0];
  const unsigned char* mask = (const unsigned char*)d_in[1];  // bool msa_mask (B,S,I)
  const float*         lnw  = (const float*)d_in[2];
  const float*         lnb  = (const float*)d_in[3];
  const float*         Wq   = (const float*)d_in[4];
  const float*         Wk   = (const float*)d_in[5];
  const float*         Wv   = (const float*)d_in[6];
  const float*         Wg   = (const float*)d_in[7];
  const float*         bg   = (const float*)d_in[8];
  const float*         Wo   = (const float*)d_in[9];
  const float*         bo   = (const float*)d_in[10];
  float* out = (float*)d_out;

  float* ws = (float*)d_ws;
  const size_t HSZ = (size_t)I_LEN * NHEAD * S_LEN * CHD;    // 3,145,728 floats per tensor
  float* q = ws, *k = ws + HSZ, *v = ws + 2*HSZ, *gate = ws + 3*HSZ, *o = ws + 4*HSZ;

  dim3 blk(256);
  const int tiles_blocks = NPOS / 16 / 8;                    // 768 blocks, 8 waves each
  k_ln_qkvg<<<tiles_blocks, blk, 0, stream>>>(m, lnw, lnb, Wq, Wk, Wv, Wg, bg, q, k, v, gate);

  const size_t smem = 12288                                  // kbuf/qbuf/vbuf (f32)
                    + (size_t)8 * 16 * S_LEN * sizeof(float); // 8 score strips = 128 KB
  k_attn<<<I_LEN * NHEAD, blk, smem, stream>>>(q, k, v, mask, o);

  k_gate_out<<<tiles_blocks, blk, 0, stream>>>(o, gate, Wo, bo, out);
}
